// GRUPlant_88536455840161
// MI455X (gfx1250) — matrix operand order repaired and checked by reading, not compiled, not run on hardware
//
#include <hip/hip_runtime.h>
#include <hip/hip_bf16.h>
#include <stdint.h>
#include <stddef.h>

typedef __bf16 bf16;
typedef __attribute__((ext_vector_type(16))) __bf16 v16bf;
typedef __attribute__((ext_vector_type(8)))  __bf16 v8bf;
typedef __attribute__((ext_vector_type(8)))  float  v8f;

#define B_   256
#define S_   256
#define T_   256
#define H_   512
#define G3   1536   // 3*H
#define NWG  96
#define NTHR 256

// ---------------------------------------------------------------- utilities
__device__ __forceinline__ float sigmoidf_(float x) {
    return 1.0f / (1.0f + __expf(-x));
}
__device__ __forceinline__ float tanhf_(float x) {
    float e2 = __expf(2.0f * x);
    return (e2 - 1.0f) / (e2 + 1.0f);
}

// Device-wide barrier for persistent kernels (atomic counter + generation).
__device__ __forceinline__ void gridbar(int* cnt, int& gen) {
    __syncthreads();
    if (threadIdx.x == 0) {
        gen += (int)gridDim.x;
        __hip_atomic_fetch_add(cnt, 1, __ATOMIC_ACQ_REL, __HIP_MEMORY_SCOPE_AGENT);
        while (__hip_atomic_load(cnt, __ATOMIC_ACQUIRE, __HIP_MEMORY_SCOPE_AGENT) < gen)
            __builtin_amdgcn_s_sleep(1);
    }
    __syncthreads();
}

// ------------------------------------------------------- WMMA tile GEMM core
// C[M=16*mT, N=16*nT] = A[M, 32*kT] * Bpacked + bias, per-wave 16x16 tiles.
// A: row-major bf16, lda elements per row (16B-aligned rows).
// Bpacked: lane-contiguous tiles, element e of lane l in tile (kt,nt) holds
//   W[nt*16 + (l&15)][kt*32 + ((l&16)?8:0) + (e<8 ? e : e+8)]   (B 32x16 ISA layout)
// MODE 0: Cf[m*ldc+g] = acc + bias[g]            (f32 gate buffers)
// MODE 1: Cb[m*ldc+g] = bf16(relu(acc+bias[g]))  (FC hidden)
// MODE 2: g<5: out[(m*T_+t)*5+g] = pv[m*5+g] = acc + bias[g]
template <int MODE>
__device__ void gemm16(const bf16* __restrict__ A, int lda,
                       const bf16* __restrict__ Bp,
                       int mT, int nT, int kT,
                       const float* __restrict__ bias,
                       float* __restrict__ Cf, int ldc,
                       bf16* __restrict__ Cb,
                       float* __restrict__ out2, float* __restrict__ pv, int t)
{
    const int lane   = threadIdx.x & 31;
    const int wave   = (int)((blockIdx.x * blockDim.x + threadIdx.x) >> 5);
    const int nWaves = (int)((gridDim.x * blockDim.x) >> 5);
    const int r      = lane & 15;
    const int koA    = (lane & 16) ? 8 : 0;   // 16-bit A 16x32 lane K-offset
    const int total  = mT * nT;

    for (int tile = wave; tile < total; tile += nWaves) {
        const int mt = tile / nT;
        const int nt = tile - mt * nT;
        v8f acc = {};
        const bf16* arow = A + (size_t)(mt * 16 + r) * lda + koA;
        for (int kt = 0; kt < kT; ++kt) {
            const bf16* pa = arow + kt * 32;
            v8bf alo = *(const v8bf*)pa;            // K = base .. base+7
            v8bf ahi = *(const v8bf*)(pa + 16);     // K = base+16 .. base+23
            v16bf a;
#pragma unroll
            for (int e = 0; e < 8; ++e) { a[e] = alo[e]; a[e + 8] = ahi[e]; }
            v16bf b = *(const v16bf*)(Bp + ((size_t)(kt * nT + nt) * 32 + lane) * 16);
            acc = __builtin_amdgcn_wmma_f32_16x16x32_bf16(
                false, a, false, b, (short)0, acc, false, false);
        }
        // C layout: lane gives N (lane&15); VGPR v gives M = v + (lane>=16 ? 8:0)
        const int n  = lane & 15;
        const int mo = (lane & 16) ? 8 : 0;
        const int g  = nt * 16 + n;
        if constexpr (MODE == 0) {
            const float bs = bias[g];
#pragma unroll
            for (int v = 0; v < 8; ++v)
                Cf[(size_t)(mt * 16 + v + mo) * ldc + g] = acc[v] + bs;
        } else if constexpr (MODE == 1) {
            const float bs = bias[g];
#pragma unroll
            for (int v = 0; v < 8; ++v) {
                float x = acc[v] + bs;
                Cb[(size_t)(mt * 16 + v + mo) * ldc + g] = (bf16)(x > 0.0f ? x : 0.0f);
            }
        } else {
            if (g < 5) {
                const float bs = bias[g];
#pragma unroll
                for (int v = 0; v < 8; ++v) {
                    const int m = mt * 16 + v + mo;
                    float x = acc[v] + bs;
                    out2[((size_t)m * T_ + t) * 5 + g] = x;
                    pv[m * 5 + g] = x;
                }
            }
        }
    }
}

// ------------------------------------------------------------- gate update
__device__ void gru_gates_phase(const float* __restrict__ gx,
                                const float* __restrict__ gh,
                                float* __restrict__ hf, bf16* __restrict__ hb)
{
    const int gsz = (int)(gridDim.x * blockDim.x);
    for (int i = (int)(blockIdx.x * blockDim.x + threadIdx.x); i < B_ * H_; i += gsz) {
        const int b = i >> 9;           // /512
        const int k = i & (H_ - 1);
        const float* px = gx + (size_t)b * G3;
        const float* ph = gh + (size_t)b * G3;
        float rg = sigmoidf_(px[k] + ph[k]);
        float zg = sigmoidf_(px[H_ + k] + ph[H_ + k]);
        float ng = tanhf_(px[2 * H_ + k] + rg * ph[2 * H_ + k]);
        float h  = (1.0f - zg) * ng + zg * hf[i];
        hf[i] = h;
        hb[i] = (bf16)h;
    }
}

// ----------------------------------------------------------- weight packing
// W is f32 [Nreal][Kreal] row-major (ldw). Output: bf16 B-tiles, zero padded.
__global__ void pack_b_kernel(const float* __restrict__ W, int Kreal, int ldw,
                              int Nreal, int kT, int nT, bf16* __restrict__ out)
{
    const int idx = (int)(blockIdx.x * blockDim.x + threadIdx.x);
    const int total = kT * nT * 32;
    if (idx >= total) return;
    const int lane = idx & 31;
    const int tile = idx >> 5;
    const int kt = tile / nT;
    const int nt = tile - kt * nT;
    const int n  = nt * 16 + (lane & 15);
    const int kb = kt * 32 + ((lane & 16) ? 16 : 0);
    bf16* dst = out + (size_t)idx * 16;
#pragma unroll
    for (int e = 0; e < 16; ++e) {
        const int k = kt * 32 + ((lane & 16) ? 8 : 0) + (e < 8 ? e : e + 8);
        float v = (n < Nreal && k < Kreal) ? W[(size_t)n * ldw + k] : 0.0f;
        dst[e] = (bf16)v;
    }
}

// --------------------------------------------------------------- init state
__global__ void init_kernel(const float* __restrict__ pv_init,
                            const int* __restrict__ scen,
                            const float* __restrict__ emb,
                            float* h0f, float* h1f, bf16* h0b, bf16* h1b,
                            bf16* e_bf, float* pv, int* cnt)
{
    const int i   = (int)(blockIdx.x * blockDim.x + threadIdx.x);
    const int gsz = (int)(gridDim.x * blockDim.x);
    for (int j = i; j < B_ * H_; j += gsz) {
        h0f[j] = 0.0f; h1f[j] = 0.0f;
        h0b[j] = (bf16)0.0f; h1b[j] = (bf16)0.0f;
    }
    for (int j = i; j < B_ * 32; j += gsz) {
        const int b = j >> 5, k = j & 31;
        e_bf[j] = (bf16)emb[scen[b] * 32 + k];
    }
    for (int j = i; j < B_ * 5; j += gsz) pv[j] = pv_init[j];
    if (i < 2) cnt[i] = 0;
}

// ------------------------------------------------------------------ encoder
struct EncArgs {
    const float* x_cv;            // [B,S,128]
    const bf16*  e_bf;            // [B,32]
    const bf16  *pWih0, *pWhh0, *pWih1, *pWhh1;
    const float *bih0, *bhh0, *bih1, *bhh1;
    bf16*  inp;                   // [B,160]
    float *gx, *gh;               // [B,1536]
    float *h0f, *h1f;
    bf16  *h0b, *h1b;
    int*   cnt;
};

__device__ void build_enc_inp(const EncArgs& a, int t) {
    const int gsz = (int)(gridDim.x * blockDim.x);
    for (int i = (int)(blockIdx.x * blockDim.x + threadIdx.x); i < B_ * 160; i += gsz) {
        const int b = i / 160, k = i - b * 160;
        bf16 v;
        if (k < 128) v = (bf16)a.x_cv[((size_t)b * S_ + t) * 128 + k];
        else         v = a.e_bf[b * 32 + (k - 128)];
        a.inp[i] = v;
    }
}

__global__ void enc_kernel(EncArgs a) {
    int gen = 0;
    build_enc_inp(a, 0);
    gridbar(a.cnt, gen);
    for (int t = 0; t < S_; ++t) {
        gemm16<0>(a.inp, 160, a.pWih0, 16, 96, 5,  a.bih0, a.gx, G3, nullptr, nullptr, nullptr, 0);
        gemm16<0>(a.h0b, 512, a.pWhh0, 16, 96, 16, a.bhh0, a.gh, G3, nullptr, nullptr, nullptr, 0);
        gridbar(a.cnt, gen);
        gru_gates_phase(a.gx, a.gh, a.h0f, a.h0b);
        gridbar(a.cnt, gen);
        gemm16<0>(a.h0b, 512, a.pWih1, 16, 96, 16, a.bih1, a.gx, G3, nullptr, nullptr, nullptr, 0);
        gemm16<0>(a.h1b, 512, a.pWhh1, 16, 96, 16, a.bhh1, a.gh, G3, nullptr, nullptr, nullptr, 0);
        gridbar(a.cnt, gen);
        gru_gates_phase(a.gx, a.gh, a.h1f, a.h1b);
        if (t + 1 < S_) build_enc_inp(a, t + 1);
        gridbar(a.cnt, gen);
    }
}

// ------------------------------------------------------------------ decoder
struct DecArgs {
    const float* x_t;             // [B,T,128]
    const bf16  *pWih0, *pWhh0, *pWih1, *pWhh1, *pW1, *pW2;
    const float *bih0, *bhh0, *bih1, *bhh1, *b1, *b2;
    bf16*  inp;                   // [B,160] (K padded: 133 -> 160)
    float *gx, *gh;
    float *h0f, *h1f;             // initialized to encoder hT0/hT1 (same bufs)
    bf16  *h0b, *h1b;
    bf16*  fc1b;                  // [B,128]
    float* pv;                    // [B,5] feedback
    float* out;                   // [B,T,5]
    int*   cnt;
};

__device__ void build_dec_inp(const DecArgs& a, int t) {
    const int gsz = (int)(gridDim.x * blockDim.x);
    for (int i = (int)(blockIdx.x * blockDim.x + threadIdx.x); i < B_ * 160; i += gsz) {
        const int b = i / 160, k = i - b * 160;
        bf16 v;
        if      (k < 128) v = (bf16)a.x_t[((size_t)b * T_ + t) * 128 + k];
        else if (k < 133) v = (bf16)a.pv[b * 5 + (k - 128)];
        else              v = (bf16)0.0f;
        a.inp[i] = v;
    }
}

__global__ void dec_kernel(DecArgs a) {
    int gen = 0;
    for (int t = 0; t < T_; ++t) {
        build_dec_inp(a, t);
        gridbar(a.cnt, gen);
        gemm16<0>(a.inp, 160, a.pWih0, 16, 96, 5,  a.bih0, a.gx, G3, nullptr, nullptr, nullptr, 0);
        gemm16<0>(a.h0b, 512, a.pWhh0, 16, 96, 16, a.bhh0, a.gh, G3, nullptr, nullptr, nullptr, 0);
        gridbar(a.cnt, gen);
        gru_gates_phase(a.gx, a.gh, a.h0f, a.h0b);
        gridbar(a.cnt, gen);
        gemm16<0>(a.h0b, 512, a.pWih1, 16, 96, 16, a.bih1, a.gx, G3, nullptr, nullptr, nullptr, 0);
        gemm16<0>(a.h1b, 512, a.pWhh1, 16, 96, 16, a.bhh1, a.gh, G3, nullptr, nullptr, nullptr, 0);
        gridbar(a.cnt, gen);
        gru_gates_phase(a.gx, a.gh, a.h1f, a.h1b);
        gridbar(a.cnt, gen);
        // fc1 = relu(h1 @ W1^T + b1) -> bf16 [B,128]
        gemm16<1>(a.h1b, 512, a.pW1, 16, 8, 16, a.b1, nullptr, 128, a.fc1b, nullptr, nullptr, 0);
        gridbar(a.cnt, gen);
        // pv_pred = fc1 @ W2^T + b2 -> out[:,t,:] and feedback pv
        gemm16<2>(a.fc1b, 128, a.pW2, 16, 1, 4, a.b2, nullptr, 0, nullptr, a.out, a.pv, t);
        gridbar(a.cnt, gen);
    }
}

// -------------------------------------------------------------------- host
extern "C" void kernel_launch(void* const* d_in, const int* in_sizes, int n_in,
                              void* d_out, int out_size, void* d_ws, size_t ws_size,
                              hipStream_t stream)
{
    (void)in_sizes; (void)n_in; (void)out_size; (void)ws_size;

    const float* x_cv   = (const float*)d_in[0];
    const float* x_tgt  = (const float*)d_in[1];
    const float* pv0    = (const float*)d_in[2];
    const int*   scen   = (const int*)d_in[3];
    const float* emb    = (const float*)d_in[4];
    const float* Wih_e0 = (const float*)d_in[5];
    const float* Whh_e0 = (const float*)d_in[6];
    const float* bih_e0 = (const float*)d_in[7];
    const float* bhh_e0 = (const float*)d_in[8];
    const float* Wih_e1 = (const float*)d_in[9];
    const float* Whh_e1 = (const float*)d_in[10];
    const float* bih_e1 = (const float*)d_in[11];
    const float* bhh_e1 = (const float*)d_in[12];
    const float* Wih_d0 = (const float*)d_in[13];
    const float* Whh_d0 = (const float*)d_in[14];
    const float* bih_d0 = (const float*)d_in[15];
    const float* bhh_d0 = (const float*)d_in[16];
    const float* Wih_d1 = (const float*)d_in[17];
    const float* Whh_d1 = (const float*)d_in[18];
    const float* bih_d1 = (const float*)d_in[19];
    const float* bhh_d1 = (const float*)d_in[20];
    const float* W1     = (const float*)d_in[21];
    const float* b1     = (const float*)d_in[22];
    const float* W2     = (const float*)d_in[23];
    const float* b2     = (const float*)d_in[24];
    float* out = (float*)d_out;

    // bump allocator in workspace (256B aligned)
    char* base = (char*)d_ws;
    size_t off = 0;
    auto alloc = [&](size_t bytes) -> void* {
        void* p = base + off;
        off = (off + bytes + 255) & ~(size_t)255;
        return p;
    };
    int*  cnt     = (int*)alloc(256);
    bf16* pWih_e0 = (bf16*)alloc((size_t)5  * 96 * 512 * 2);
    bf16* pWhh_e0 = (bf16*)alloc((size_t)16 * 96 * 512 * 2);
    bf16* pWih_e1 = (bf16*)alloc((size_t)16 * 96 * 512 * 2);
    bf16* pWhh_e1 = (bf16*)alloc((size_t)16 * 96 * 512 * 2);
    bf16* pWih_d0 = (bf16*)alloc((size_t)5  * 96 * 512 * 2);
    bf16* pWhh_d0 = (bf16*)alloc((size_t)16 * 96 * 512 * 2);
    bf16* pWih_d1 = (bf16*)alloc((size_t)16 * 96 * 512 * 2);
    bf16* pWhh_d1 = (bf16*)alloc((size_t)16 * 96 * 512 * 2);
    bf16* pW1     = (bf16*)alloc((size_t)16 * 8  * 512 * 2);
    bf16* pW2     = (bf16*)alloc((size_t)4  * 1  * 512 * 2);
    bf16* e_bf    = (bf16*)alloc((size_t)B_ * 32 * 2);
    bf16* inp     = (bf16*)alloc((size_t)B_ * 160 * 2);
    float* gx     = (float*)alloc((size_t)B_ * G3 * 4);
    float* gh     = (float*)alloc((size_t)B_ * G3 * 4);
    float* h0f    = (float*)alloc((size_t)B_ * H_ * 4);
    float* h1f    = (float*)alloc((size_t)B_ * H_ * 4);
    bf16*  h0b    = (bf16*)alloc((size_t)B_ * H_ * 2);
    bf16*  h1b    = (bf16*)alloc((size_t)B_ * H_ * 2);
    bf16*  fc1b   = (bf16*)alloc((size_t)B_ * 128 * 2);
    float* pv     = (float*)alloc((size_t)B_ * 5 * 4);

    auto pack = [&](const float* W, int Kreal, int ldw, int Nreal,
                    int kT, int nT, bf16* dst) {
        int total = kT * nT * 32;
        pack_b_kernel<<<(total + 255) / 256, 256, 0, stream>>>(
            W, Kreal, ldw, Nreal, kT, nT, dst);
    };
    pack(Wih_e0, 160, 160, G3, 5,  96, pWih_e0);
    pack(Whh_e0, 512, 512, G3, 16, 96, pWhh_e0);
    pack(Wih_e1, 512, 512, G3, 16, 96, pWih_e1);
    pack(Whh_e1, 512, 512, G3, 16, 96, pWhh_e1);
    pack(Wih_d0, 133, 133, G3, 5,  96, pWih_d0);
    pack(Whh_d0, 512, 512, G3, 16, 96, pWhh_d0);
    pack(Wih_d1, 512, 512, G3, 16, 96, pWih_d1);
    pack(Whh_d1, 512, 512, G3, 16, 96, pWhh_d1);
    pack(W1,     512, 512, 128, 16, 8, pW1);
    pack(W2,     128, 128, 5,   4,  1, pW2);

    init_kernel<<<128, 256, 0, stream>>>(pv0, scen, emb, h0f, h1f, h0b, h1b,
                                         e_bf, pv, cnt);

    EncArgs ea;
    ea.x_cv = x_cv; ea.e_bf = e_bf;
    ea.pWih0 = pWih_e0; ea.pWhh0 = pWhh_e0; ea.pWih1 = pWih_e1; ea.pWhh1 = pWhh_e1;
    ea.bih0 = bih_e0; ea.bhh0 = bhh_e0; ea.bih1 = bih_e1; ea.bhh1 = bhh_e1;
    ea.inp = inp; ea.gx = gx; ea.gh = gh;
    ea.h0f = h0f; ea.h1f = h1f; ea.h0b = h0b; ea.h1b = h1b;
    ea.cnt = cnt + 0;
    enc_kernel<<<NWG, NTHR, 0, stream>>>(ea);

    DecArgs da;
    da.x_t = x_tgt;
    da.pWih0 = pWih_d0; da.pWhh0 = pWhh_d0; da.pWih1 = pWih_d1; da.pWhh1 = pWhh_d1;
    da.pW1 = pW1; da.pW2 = pW2;
    da.bih0 = bih_d0; da.bhh0 = bhh_d0; da.bih1 = bih_d1; da.bhh1 = bhh_d1;
    da.b1 = b1; da.b2 = b2;
    da.inp = inp; da.gx = gx; da.gh = gh;
    da.h0f = h0f; da.h1f = h1f; da.h0b = h0b; da.h1b = h1b;  // hT0/hT1 carry over
    da.fc1b = fc1b; da.pv = pv; da.out = out;
    da.cnt = cnt + 1;
    dec_kernel<<<NWG, NTHR, 0, stream>>>(da);
}
